// Occulsion_Aware_FeatureExtractor_62526133895474
// MI455X (gfx1250) — matrix operand order repaired and checked by reading, not compiled, not run on hardware
//
#include <hip/hip_runtime.h>
#include <hip/hip_bf16.h>
#include <math.h>

typedef __attribute__((ext_vector_type(16))) _Float16 v16h;
typedef __attribute__((ext_vector_type(8)))  float    v8f;

#define BLOCK_THREADS 128   // 4 waves (wave32)

__device__ __forceinline__ int iclamp(int v, int lo, int hi) {
  return v < lo ? lo : (v > hi ? hi : v);
}

// ---------------------------------------------------------------------------
// Implicit-GEMM 3x3 conv (pad=1), NCHW f32 in/out, f16 WMMA math, f32 accum.
// Per block: 16 output channels x 128 pixels (4 waves, 32 pixels/wave, two
// 16x16 wmma tiles per wave sharing one LDS-staged A fragment).
// Requires Cout % 16 == 0 and (Hout*Wout) % 128 == 0 (true for all layers).
//
// ALIGNED (Cin % 32 == 0): tap-major K order -> each 32-wide K chunk sits in
// one 3x3 tap; spatial offset/mask loop-invariant per tap; double-buffered
// LDS weight staging (one barrier per chunk, staging overlapped with wmma).
// Generic path: dense k = ci*9+tap with clamp+mask tail, simple staging.
// All predication is branch-free (clamped addresses + 0/1 multiply).
// ---------------------------------------------------------------------------
template <bool ALIGNED>
__global__ __launch_bounds__(BLOCK_THREADS) void conv3x3_wmma_kernel(
    const float* __restrict__ in, const float* __restrict__ wgt,
    const float* __restrict__ bias, float* __restrict__ out,
    int Cin, int Cout, int Hin, int Win, int Hout, int Wout,
    int stride, long inBS, long outBS, int silu)
{
  __shared__ alignas(16) _Float16 sA[2][512];   // ping-pong 16x32 f16 tiles

  const int tid  = threadIdx.x;
  const int lane = tid & 31;
  const int b    = blockIdx.z;
  const int co0  = blockIdx.y << 4;
  const int Np   = Hout * Wout;
  const int p0   = blockIdx.x * 128 + (tid >> 5) * 32;

  const float* inb = in + (long)b * inBS;
  const int K    = Cin * 9;
  const int HWin = Hin * Win;

  // A fragment (16-bit 16x32): lane row = lane&15; lanes 0-15 hold K 0-15,
  // lanes 16-31 hold K 16-31.
  const int mrow = lane & 15;
  const int akb  = (lane < 16) ? 0 : 16;
  // B fragment (32x16): lane column = lane&15; K half selected by lane group.
  const int bkb  = (lane < 16) ? 0 : 16;
  const int pA   = p0 + (lane & 15);
  const int pB   = pA + 16;
  const int oyA  = pA / Wout, oxA = pA - oyA * Wout;
  const int oyB  = pB / Wout, oxB = pB - oyB * Wout;

  // Per-thread A-staging coordinates (4 elements each).
  const int stM = (tid * 4) >> 5;        // row m (0..15)
  const int stK = (tid * 4) & 31;        // k within chunk (0,4,8,...,28)

  v8f acc0 = {}, acc1 = {};

  if constexpr (ALIGNED) {
    const int nblk = Cin >> 5;

    // Prologue: stage (tap=0, c0=0) into buffer 0.
#pragma unroll
    for (int q = 0; q < 4; ++q)
      sA[0][tid * 4 + q] =
          (_Float16)wgt[(long)(co0 + stM) * K + (long)(stK + q) * 9 + 0];
    __syncthreads();

    int bufsel = 0;
    for (int tap = 0; tap < 9; ++tap) {
      const int ky = tap / 3;
      const int kx = tap - ky * 3;
      const int iyA = oyA * stride + ky - 1, ixA = oxA * stride + kx - 1;
      const int iyB = oyB * stride + ky - 1, ixB = oxB * stride + kx - 1;
      const float mskA = (iyA >= 0 && iyA < Hin && ixA >= 0 && ixA < Win) ? 1.0f : 0.0f;
      const float mskB = (iyB >= 0 && iyB < Hin && ixB >= 0 && ixB < Win) ? 1.0f : 0.0f;
      const long spA = (long)iclamp(iyA, 0, Hin - 1) * Win + iclamp(ixA, 0, Win - 1);
      const long spB = (long)iclamp(iyB, 0, Hin - 1) * Win + iclamp(ixB, 0, Win - 1);

      for (int cb = 0; cb < nblk; ++cb) {
        const int c0 = cb << 5;

        // Read current A fragment from LDS (two b128 loads per lane).
        v16h a;
#pragma unroll
        for (int j = 0; j < 8; ++j) {
          a[j]     = sA[bufsel][mrow * 32 + akb + j];
          a[8 + j] = sA[bufsel][mrow * 32 + 8 + akb + j];
        }

        // Stage next chunk into the other buffer (overlaps with wmma below).
        {
          int c0N = c0 + 32, tapN = tap;
          if (c0N == Cin) { c0N = 0; ++tapN; }
          if (tapN < 9) {
#pragma unroll
            for (int q = 0; q < 4; ++q)
              sA[bufsel ^ 1][tid * 4 + q] =
                  (_Float16)wgt[(long)(co0 + stM) * K + (long)(c0N + stK + q) * 9 + tapN];
          }
        }

        // B gather: issue all raw loads first, then convert.
        const float* qch = inb + (long)(c0 + bkb) * HWin;
        float rA[16], rB[16];
#pragma unroll
        for (int j = 0; j < 16; ++j) rA[j] = qch[(long)j * HWin + spA];
#pragma unroll
        for (int j = 0; j < 16; ++j) rB[j] = qch[(long)j * HWin + spB];
        v16h b0, b1;
#pragma unroll
        for (int j = 0; j < 16; ++j) b0[j] = (_Float16)(rA[j] * mskA);
#pragma unroll
        for (int j = 0; j < 16; ++j) b1[j] = (_Float16)(rB[j] * mskB);

        acc0 = __builtin_amdgcn_wmma_f32_16x16x32_f16(
            false, a, false, b0, (short)0, acc0, false, false);
        acc1 = __builtin_amdgcn_wmma_f32_16x16x32_f16(
            false, a, false, b1, (short)0, acc1, false, false);

        __syncthreads();   // staging of next buffer complete before next read
        bufsel ^= 1;
      }
    }
  } else {
    const int nchunks = (K + 31) >> 5;
    for (int chb = 0; chb < nchunks; ++chb) {
      const int k0 = chb << 5;
#pragma unroll
      for (int q = 0; q < 4; ++q) {
        const int kk = k0 + stK + q;
        const int kc = kk < K ? kk : K - 1;
        const float wv = wgt[(long)(co0 + stM) * K + kc];
        sA[0][tid * 4 + q] = (_Float16)(kk < K ? wv : 0.0f);
      }
      __syncthreads();

      v16h a;
#pragma unroll
      for (int j = 0; j < 8; ++j) {
        a[j]     = sA[0][mrow * 32 + akb + j];
        a[8 + j] = sA[0][mrow * 32 + 8 + akb + j];
      }
      float rA[16], rB[16];
      float mA[16], mB[16];
#pragma unroll
      for (int j = 0; j < 16; ++j) {
        const int kk = k0 + bkb + j;
        const int kc = kk < K ? kk : 0;
        const int ci = kc / 9;
        const int tap = kc - ci * 9;
        const int ky = tap / 3, kx = tap - ky * 3;
        const int iyA = oyA * stride + ky - 1, ixA = oxA * stride + kx - 1;
        const int iyB = oyB * stride + ky - 1, ixB = oxB * stride + kx - 1;
        mA[j] = (kk < K && iyA >= 0 && iyA < Hin && ixA >= 0 && ixA < Win) ? 1.0f : 0.0f;
        mB[j] = (kk < K && iyB >= 0 && iyB < Hin && ixB >= 0 && ixB < Win) ? 1.0f : 0.0f;
        const long adA = (long)ci * HWin + (long)iclamp(iyA, 0, Hin - 1) * Win + iclamp(ixA, 0, Win - 1);
        const long adB = (long)ci * HWin + (long)iclamp(iyB, 0, Hin - 1) * Win + iclamp(ixB, 0, Win - 1);
        rA[j] = inb[adA];
        rB[j] = inb[adB];
      }
      v16h b0, b1;
#pragma unroll
      for (int j = 0; j < 16; ++j) b0[j] = (_Float16)(rA[j] * mA[j]);
#pragma unroll
      for (int j = 0; j < 16; ++j) b1[j] = (_Float16)(rB[j] * mB[j]);

      acc0 = __builtin_amdgcn_wmma_f32_16x16x32_f16(
          false, a, false, b0, (short)0, acc0, false, false);
      acc1 = __builtin_amdgcn_wmma_f32_16x16x32_f16(
          false, a, false, b1, (short)0, acc1, false, false);
      __syncthreads();
    }
  }

  // Epilogue: lane column = lane&15, row = r + 8*(lane>=16). No guards needed.
#pragma unroll
  for (int r = 0; r < 8; ++r) {
    const int co = co0 + r + ((lane < 16) ? 0 : 8);
    const float bb = bias[co];
    float v0 = acc0[r] + bb;
    float v1 = acc1[r] + bb;
    if (silu) {
      v0 = v0 / (1.0f + __expf(-v0));
      v1 = v1 / (1.0f + __expf(-v1));
    }
    float* op = out + (long)b * outBS + (long)co * Np;
    op[pA] = v0;
    op[pB] = v1;
  }
}

// ---------------------------------------------------------------------------
// Bilinear resize of one flow pair (B,2,512,512) -> (B,2,r,r), * 2/(r-1).
// ---------------------------------------------------------------------------
__global__ void resize_flow_kernel(const float* __restrict__ flow, int chanOff,
                                   float* __restrict__ outp, int Hin, int r,
                                   float norm, int B)
{
  int idx = blockIdx.x * blockDim.x + threadIdx.x;
  const int rr = r * r;
  if (idx >= B * 2 * rr) return;
  const int p  = idx % rr;
  const int t  = idx / rr;
  const int cc = t & 1;
  const int b  = t >> 1;
  const int y  = p / r;
  const int x  = p - y * r;
  const float sc = (float)Hin / (float)r;
  const float fy = ((float)y + 0.5f) * sc - 0.5f;
  const float fx = ((float)x + 0.5f) * sc - 0.5f;
  const float y0f = floorf(fy), x0f = floorf(fx);
  const float ty = fy - y0f, tx = fx - x0f;
  int y0 = iclamp((int)y0f, 0, Hin - 1);
  int y1 = iclamp((int)y0f + 1, 0, Hin - 1);
  int x0 = iclamp((int)x0f, 0, Hin - 1);
  int x1 = iclamp((int)x0f + 1, 0, Hin - 1);
  const float* f = flow + ((long)b * 4 + chanOff + cc) * (long)Hin * Hin;
  const float v00 = f[(long)y0 * Hin + x0];
  const float v01 = f[(long)y0 * Hin + x1];
  const float v10 = f[(long)y1 * Hin + x0];
  const float v11 = f[(long)y1 * Hin + x1];
  const float v = (1.0f - ty) * ((1.0f - tx) * v00 + tx * v01)
                + ty * ((1.0f - tx) * v10 + tx * v11);
  outp[idx] = v * norm;
}

// ---------------------------------------------------------------------------
// Forward splat: acc[b, c] += (c<C ? src*e : e) * bilinear weight at flow dst.
// Thread per (b, c, source pixel). metric == ones -> m = e.
// ---------------------------------------------------------------------------
__global__ void splat_kernel(const float* __restrict__ src, long srcBS,
                             const float* __restrict__ fl, float* __restrict__ acc,
                             int B, int C, int r)
{
  long idx = (long)blockIdx.x * blockDim.x + threadIdx.x;
  const long rr = (long)r * r;
  const long total = (long)B * (C + 1) * rr;
  if (idx >= total) return;
  const int p = (int)(idx % rr);
  long t = idx / rr;
  const int c = (int)(t % (C + 1));
  const int b = (int)(t / (C + 1));
  const int y = p / r;
  const int x = p - y * r;
  const float* f = fl + (long)b * 2 * rr;
  const float gx = (float)x + f[p];
  const float gy = (float)y + f[rr + p];
  const float x0f = floorf(gx), y0f = floorf(gy);
  const int x0 = (int)x0f, y0 = (int)y0f;
  const int x1 = x0 + 1,  y1 = y0 + 1;
  const float wx1 = gx - x0f, wx0 = 1.0f - wx1;
  const float wy1 = gy - y0f, wy0 = 1.0f - wy1;
  const float M = 2.7182818284590452f;           // exp(metric = 1)
  const float val = (c < C) ? src[(long)b * srcBS + (long)c * rr + p] * M : M;
  float* ab = acc + ((long)b * (C + 1) + c) * rr;
  if (x0 >= 0 && x0 < r && y0 >= 0 && y0 < r) atomicAdd(ab + (long)y0 * r + x0, val * wx0 * wy0);
  if (x1 >= 0 && x1 < r && y0 >= 0 && y0 < r) atomicAdd(ab + (long)y0 * r + x1, val * wx1 * wy0);
  if (x0 >= 0 && x0 < r && y1 >= 0 && y1 < r) atomicAdd(ab + (long)y1 * r + x0, val * wx0 * wy1);
  if (x1 >= 0 && x1 < r && y1 >= 0 && y1 < r) atomicAdd(ab + (long)y1 * r + x1, val * wx1 * wy1);
}

// ---------------------------------------------------------------------------
// occ = (|flow_b + warped| > 0.3), warped = acc[0:2]/(acc[2]+1e-7).
// ---------------------------------------------------------------------------
__global__ void mask_kernel(const float* __restrict__ acc, const float* __restrict__ flB,
                            float* __restrict__ occ, int B, int r)
{
  int idx = blockIdx.x * blockDim.x + threadIdx.x;
  const int rr = r * r;
  if (idx >= B * rr) return;
  const int b = idx / rr;
  const int p = idx - b * rr;
  const float* a = acc + (long)b * 3 * rr;
  const float d  = a[2 * rr + p] + 1e-7f;
  const float* f = flB + (long)b * 2 * rr;
  const float dx = f[p] + a[p] / d;
  const float dy = f[rr + p] + a[rr + p] / d;
  occ[idx] = (sqrtf(dx * dx + dy * dy) > 0.3f) ? 1.0f : 0.0f;
}

// ---------------------------------------------------------------------------
// cat[.., catOff + c] = warped*(1-occ) + feat*occ, warped = acc[c]/(acc[C]+eps)
// ---------------------------------------------------------------------------
__global__ void warp_combine_kernel(const float* __restrict__ acc, const float* __restrict__ feat,
                                    const float* __restrict__ occ, float* __restrict__ cat,
                                    int B, int C, int r, int catC, int catOff)
{
  long idx = (long)blockIdx.x * blockDim.x + threadIdx.x;
  const long rr = (long)r * r;
  const long total = (long)B * C * rr;
  if (idx >= total) return;
  const int p = (int)(idx % rr);
  long t = idx / rr;
  const int c = (int)(t % C);
  const int b = (int)(t / C);
  const float* ab = acc + (long)b * (C + 1) * rr;
  const float d = ab[(long)C * rr + p] + 1e-7f;
  const float w = ab[(long)c * rr + p] / d;
  const float o = occ[(long)b * rr + p];
  const float fv = feat[((long)b * C + c) * rr + p];
  cat[((long)b * catC + catOff + c) * rr + p] = w * (1.0f - o) + fv * o;
}

// ===========================================================================
extern "C" void kernel_launch(void* const* d_in, const int* in_sizes, int n_in,
                              void* d_out, int out_size, void* d_ws, size_t ws_size,
                              hipStream_t stream) {
  (void)out_size; (void)ws_size;
  const float* lc   = (const float*)d_in[0];   // (4,6,512,512)
  const float* flow = (const float*)d_in[1];   // (4,4,512,512)
  float* out = (float*)d_out;
  const int B = 4;

  // Locate weight tensors by their unique flat sizes ("first" precedes
  // "last" in both dict-insertion and pytree-sorted flatten orders).
  auto findw = [&](int sz, int occ) -> int {
    int seen = 0;
    for (int i = 0; i < n_in; ++i)
      if (in_sizes[i] == sz) { if (seen == occ) return i; ++seen; }
    return 0;
  };
  const int pf1 = findw(432, 0),  pf2 = findw(4608, 0),  pf3 = findw(9216, 0);
  const int pl1 = findw(432, 1),  pl2 = findw(4608, 1),  pl3 = findw(9216, 1);
  const int ef[4] = { findw(27648,0), findw(110592,0), findw(221184,0), findw(442368,0) };
  const int el[4] = { findw(27648,1), findw(110592,1), findw(221184,1), findw(442368,1) };
  const int zw[4] = { findw(331776,0), findw(589824,0), findw(1327104,0), findw(2359296,0) };

  // Workspace layout (floats); big pre-stage buffers are aliased later.
  float* ws = (float*)d_ws;
  size_t off = 0;
  auto alloc = [&](size_t n) { float* pp = ws + off; off += n; return pp; };
  const long HW512 = 512L * 512, HW256 = 256L * 256, HW128 = 128L * 128;
  float* P1  = alloc((size_t)B * 16 * HW512);  // 67 MB
  float* P2  = alloc((size_t)B * 32 * HW256);  // 33.5 MB
  float* FF0 = alloc((size_t)B * 32 * HW256);
  float* LF0 = alloc((size_t)B * 32 * HW256);
  float* ffA = alloc((size_t)B * 96 * HW128);
  float* ffB = alloc((size_t)B * 96 * HW128);
  float* lfA = alloc((size_t)B * 96 * HW128);
  float* lfB = alloc((size_t)B * 96 * HW128);
  float* FL  = alloc((size_t)B * 2 * HW128);
  float* FB  = alloc((size_t)B * 2 * HW128);
  float* OCF = alloc((size_t)B * HW128);
  float* OCB = alloc((size_t)B * HW128);
  float* ACC = P2;   // splat accumulator (max (96+1)*4*128^2 < |P2|)
  float* CAT = P1;   // concat buffer   (max 4*192*128^2 < |P1|)

  auto conv = [&](const float* inp, int widx, float* outp, int Cin, int Cout,
                  int Hin, int Hout, int stride, long inBS, int silu) {
    const float* w  = (const float*)d_in[widx];
    const float* bi = (const float*)d_in[widx + 1];
    const int Np = Hout * Hout;
    dim3 grid(Np / 128, Cout / 16, B);
    if ((Cin & 31) == 0)
      conv3x3_wmma_kernel<true><<<grid, BLOCK_THREADS, 0, stream>>>(
          inp, w, bi, outp, Cin, Cout, Hin, Hin, Hout, Hout, stride,
          inBS, (long)Cout * Np, silu);
    else
      conv3x3_wmma_kernel<false><<<grid, BLOCK_THREADS, 0, stream>>>(
          inp, w, bi, outp, Cin, Cout, Hin, Hin, Hout, Hout, stride,
          inBS, (long)Cout * Np, silu);
  };

  // pre_extract(first = channels 3..5) and (last = channels 0..2)
  conv(lc + 3 * HW512, pf1, P1, 3, 16, 512, 512, 1, 6 * HW512, 1);
  conv(P1, pf2, P2, 16, 32, 512, 256, 2, 16 * HW512, 1);
  conv(P2, pf3, FF0, 32, 32, 256, 256, 1, 32 * HW256, 1);
  conv(lc, pl1, P1, 3, 16, 512, 512, 1, 6 * HW512, 1);
  conv(P1, pl2, P2, 16, 32, 512, 256, 2, 16 * HW512, 1);
  conv(P2, pl3, LF0, 32, 32, 256, 256, 1, 32 * HW256, 1);

  const int chain[5]  = {32, 96, 128, 192, 256};
  const int resin[4]  = {256, 128, 64, 32};
  const int resout[4] = {128, 64, 32, 16};

  float* ffPrev = FF0;
  float* lfPrev = LF0;
  long outOff = 0;
  for (int i = 0; i < 4; ++i) {
    const int r   = resout[i];
    const long rr = (long)r * r;
    const int C   = chain[i + 1];
    float* ffCur = (i & 1) ? ffB : ffA;
    float* lfCur = (i & 1) ? lfB : lfA;
    const long inBS = (long)chain[i] * resin[i] * resin[i];

    conv(ffPrev, ef[i], ffCur, chain[i], C, resin[i], r, 2, inBS, 1);
    conv(lfPrev, el[i], lfCur, chain[i], C, resin[i], r, 2, inBS, 1);

    // flow resize + normalize
    {
      const int tot = B * 2 * (int)rr;
      const float nrm = 2.0f / (float)(r - 1);
      resize_flow_kernel<<<(tot + 255) / 256, 256, 0, stream>>>(flow, 0, FL, 512, r, nrm, B);
      resize_flow_kernel<<<(tot + 255) / 256, 256, 0, stream>>>(flow, 2, FB, 512, r, nrm, B);
    }

    // occ_fwd = mask(splat(FL via FB), FB)
    hipMemsetAsync(ACC, 0, (size_t)B * 3 * rr * sizeof(float), stream);
    { long tot = (long)B * 3 * rr;
      splat_kernel<<<(unsigned)((tot + 255) / 256), 256, 0, stream>>>(FL, 2 * rr, FB, ACC, B, 2, r); }
    { int tot = B * (int)rr;
      mask_kernel<<<(tot + 255) / 256, 256, 0, stream>>>(ACC, FB, OCF, B, r); }
    // occ_bwd = mask(splat(FB via FL), FL)
    hipMemsetAsync(ACC, 0, (size_t)B * 3 * rr * sizeof(float), stream);
    { long tot = (long)B * 3 * rr;
      splat_kernel<<<(unsigned)((tot + 255) / 256), 256, 0, stream>>>(FB, 2 * rr, FL, ACC, B, 2, r); }
    { int tot = B * (int)rr;
      mask_kernel<<<(tot + 255) / 256, 256, 0, stream>>>(ACC, FL, OCB, B, r); }

    // wrapped_first -> CAT[:, 0:C]
    hipMemsetAsync(ACC, 0, (size_t)B * (C + 1) * rr * sizeof(float), stream);
    { long tot = (long)B * (C + 1) * rr;
      splat_kernel<<<(unsigned)((tot + 255) / 256), 256, 0, stream>>>(ffCur, (long)C * rr, FL, ACC, B, C, r); }
    { long tot = (long)B * C * rr;
      warp_combine_kernel<<<(unsigned)((tot + 255) / 256), 256, 0, stream>>>(ACC, ffCur, OCF, CAT, B, C, r, 2 * C, 0); }
    // wrapped_last -> CAT[:, C:2C]
    hipMemsetAsync(ACC, 0, (size_t)B * (C + 1) * rr * sizeof(float), stream);
    { long tot = (long)B * (C + 1) * rr;
      splat_kernel<<<(unsigned)((tot + 255) / 256), 256, 0, stream>>>(lfCur, (long)C * rr, FB, ACC, B, C, r); }
    { long tot = (long)B * C * rr;
      warp_combine_kernel<<<(unsigned)((tot + 255) / 256), 256, 0, stream>>>(ACC, lfCur, OCB, CAT, B, C, r, 2 * C, C); }

    // zero conv -> output slice (no activation)
    conv(CAT, zw[i], out + outOff, 2 * C, 2 * C, r, r, 1, (long)2 * C * rr, 0);
    outOff += (long)B * 2 * C * rr;

    ffPrev = ffCur;
    lfPrev = lfCur;
  }
}
